// GraphAttention_29128468201626
// MI455X (gfx1250) — matrix operand order repaired and checked by reading, not compiled, not run on hardware
//
#include <hip/hip_runtime.h>
#include <hip/hip_bf16.h>

// ---------------------------------------------------------------------------
// Problem constants (from reference setup_inputs)
// ---------------------------------------------------------------------------
#define BB   16
#define CC   64      // channels C and O
#define NN   1024    // points
#define KK   20      // neighbors
#define HH   128     // attention hidden
#define EPS  1e-5f

typedef __attribute__((ext_vector_type(16))) _Float16 v16h;
typedef __attribute__((ext_vector_type(8)))  float    v8f;
typedef __attribute__((ext_vector_type(4)))  unsigned int u32x4;
typedef __attribute__((ext_vector_type(4)))  int      i32x4;
typedef __attribute__((ext_vector_type(8)))  int      i32x8;

// ---------------------------------------------------------------------------
// WMMA helpers (CDNA5 wave32 layouts, 16x16x32 f16 -> f32)
// ---------------------------------------------------------------------------
__device__ __forceinline__ v8f wmma32(v16h a, v16h b, v8f c) {
  return __builtin_amdgcn_wmma_f32_16x16x32_f16(false, a, false, b, (short)0, c,
                                                false, false);
}

// A fragment (16x32, element(m,k) = s[m*ms + k*ks]).
__device__ __forceinline__ v16h frag_a2(const _Float16* s, int ms, int ks) {
  int lane = threadIdx.x & 31;
  int m = lane & 15, grp = lane >> 4;
  v16h a;
#pragma unroll
  for (int v = 0; v < 8; ++v) {
    int kb = grp * 16 + 2 * v;
    a[2 * v]     = s[m * ms + kb * ks];
    a[2 * v + 1] = s[m * ms + (kb + 1) * ks];
  }
  return a;
}

// A fragment converting from f32 global memory (row-major, leading dim ld)
__device__ __forceinline__ v16h frag_a_f32(const float* g, int ld) {
  int lane = threadIdx.x & 31;
  int m = lane & 15, grp = lane >> 4;
  v16h a;
#pragma unroll
  for (int v = 0; v < 8; ++v) {
    int kb = grp * 16 + 2 * v;
    a[2 * v]     = (_Float16)g[m * ld + kb];
    a[2 * v + 1] = (_Float16)g[m * ld + kb + 1];
  }
  return a;
}

// B fragment (32x16, element(k,n) = s[k*rs + n*cs]).
__device__ __forceinline__ v16h frag_b2(const _Float16* s, int rs, int cs) {
  int lane = threadIdx.x & 31;
  int n = lane & 15, grp = lane >> 4;
  v16h b;
#pragma unroll
  for (int v = 0; v < 8; ++v) {
    int kb = grp * 16 + 2 * v;
    b[2 * v]     = s[kb * rs + n * cs];
    b[2 * v + 1] = s[(kb + 1) * rs + n * cs];
  }
  return b;
}

// B fragment from LDS staged in fragment-packed order (lane's 16 halves
// contiguous) -> 2x ds_load_b128.
__device__ __forceinline__ v16h frag_b_packed(const _Float16* s) {
  int lane = threadIdx.x & 31;
  return *((const v16h*)s + lane);
}

// Fragment-packed index for a (K x 320) f16 matrix tiled into 32K x 16N blocks:
//   element (kd, col) -> block (kd>>5, col>>4), sub-block (kd>>4)&1,
//   within: (col&15)*16 + (kd&15). 20 column tiles (320 cols).
__device__ __forceinline__ int pkidx(int kd, int col) {
  return (((kd >> 5) * 20 + (col >> 4)) * 2 + ((kd >> 4) & 1)) * 256 +
         (col & 15) * 16 + (kd & 15);
}

// B fragment for the per-k consumers: column col = p*20 + k, p = lane&15.
// Packed layout makes the lane's 16 halves contiguous.
__device__ __forceinline__ v16h frag_b_pkcol(const _Float16* buf, int ks, int k) {
  int lane = threadIdx.x & 31;
  int col = (lane & 15) * KK + k;
  const _Float16* base =
      buf + (((ks * 20 + (col >> 4)) * 2 + (lane >> 4)) * 256 + (col & 15) * 16);
  return *(const v16h*)base;
}

// Packed layout for q/k (128 x 64): 4 column tiles.
__device__ __forceinline__ int qpk(int h, int c) {
  return (((h >> 5) * 4 + (c >> 4)) * 2 + ((h >> 4) & 1)) * 256 +
         (c & 15) * 16 + (h & 15);
}
// A[m][kb] = s_packed[h=k0+kb][c=mt*16+m]  (scores GEMM, K = h)
__device__ __forceinline__ v16h frag_a_qpk(const _Float16* s, int k0, int mt) {
  int lane = threadIdx.x & 31;
  int m = lane & 15, grp = lane >> 4;
  int c = mt * 16 + m;
  const _Float16* b0 = s + (((k0 >> 5) * 4 + (c >> 4)) * 2) * 256 + (c & 15) * 16;
  v16h a;
#pragma unroll
  for (int v = 0; v < 4; ++v) {
    a[2 * v]     = b0[grp * 256 + 2 * v];
    a[2 * v + 1] = b0[grp * 256 + 2 * v + 1];
  }
#pragma unroll
  for (int v = 4; v < 8; ++v) {
    a[2 * v]     = b0[grp * 256 + 2 * v];
    a[2 * v + 1] = b0[grp * 256 + 2 * v + 1];
  }
  return a;
}
// B[kb][n] = s_packed[h=k0+kb][c=nt*16+n]
__device__ __forceinline__ v16h frag_b_qpk(const _Float16* s, int k0, int nt) {
  int lane = threadIdx.x & 31;
  int n = lane & 15, grp = lane >> 4;
  int c = nt * 16 + n;
  const _Float16* base =
      s + (((k0 >> 5) * 4 + (c >> 4)) * 2 + grp) * 256 + (c & 15) * 16;
  return *(const v16h*)base;
}

__device__ __forceinline__ float bn_apply(const float* p, int nch, int c, float v) {
  float g = p[c], be = p[nch + c], mu = p[2 * nch + c], va = p[3 * nch + c];
  return (v - mu) * (g * rsqrtf(va + EPS)) + be;
}
__device__ __forceinline__ float d_lrelu(float v) { return v > 0.f ? v : 0.2f * v; }
__device__ __forceinline__ float d_relu(float v)  { return v > 0.f ? v : 0.f; }

// ---------------------------------------------------------------------------
// Raw LDS staging: Tensor Data Mover on gfx1250 (contiguous byte copy viewed
// as a 1D tensor of 8-byte elements), cooperative uint4 copy otherwise.
// Caller must __syncthreads() afterwards.
// ---------------------------------------------------------------------------
__device__ __forceinline__ void stage_lds_copy(char* smem_base, void* dst,
                                               const void* src, int bytes) {
#if defined(__gfx1250__) && __has_builtin(__builtin_amdgcn_tensor_load_to_lds)
  if (threadIdx.x < 32) {   // one wave issues the TDM op (EXEC ignored by TDM)
    unsigned lds_off = (unsigned)((const char*)dst - smem_base);
    unsigned long long ga = (unsigned long long)src;
    int n8 = bytes >> 3;                      // 8-byte elements
    u32x4 g0 = {1u,                           // count=1 valid descriptor
                lds_off,                      // lds_addr
                (unsigned)ga,                 // global_addr[31:0]
                (unsigned)((ga >> 32) & 0x01FFFFFFu) | 0x80000000u}; // type=2
    i32x8 g1 = {};
    g1[0] = 0x00030000;                       // data_size = 8B
    g1[1] = (n8 & 0xFFFF) << 16;              // tensor_dim0[15:0]
    g1[2] = ((n8 >> 16) & 0xFFFF) | (1 << 16);// tensor_dim0[31:16] | tensor_dim1=1
    g1[3] = (n8 & 0xFFFF) << 16;              // tile_dim0 = n8
    g1[4] = 1;                                // tile_dim1 = 1
    g1[5] = n8;                               // tensor_dim0_stride
    i32x4 gz = {};
#if __clang_major__ >= 23
    i32x8 gz8 = {};
    __builtin_amdgcn_tensor_load_to_lds(g0, g1, gz, gz, gz8, 0);
#else
    __builtin_amdgcn_tensor_load_to_lds(g0, g1, gz, gz, 0);
#endif
#if __has_builtin(__builtin_amdgcn_s_wait_tensorcnt)
    __builtin_amdgcn_s_wait_tensorcnt(0);
#endif
  }
#else
  const uint4* s4 = (const uint4*)src;
  uint4* d4 = (uint4*)dst;
  int n = bytes >> 4;
  for (int i = threadIdx.x; i < n; i += blockDim.x) d4[i] = s4[i];
#endif
}

// ---------------------------------------------------------------------------
// Kernel 1: per-point squared norms  sq[b,n] = sum_c x[b,c,n]^2
// ---------------------------------------------------------------------------
__global__ void sqnorm_kernel(const float* __restrict__ x, float* __restrict__ sq) {
  int i = blockIdx.x * blockDim.x + threadIdx.x;   // b*NN + n
  if (i >= BB * NN) return;
  int b = i >> 10, n = i & (NN - 1);
  const float* xb = x + (size_t)b * CC * NN + n;
  float s = 0.f;
#pragma unroll 8
  for (int c = 0; c < CC; ++c) { float v = xb[c * NN]; s += v * v; }
  sq[i] = s;
}

// ---------------------------------------------------------------------------
// Kernel 2: Gram-matrix tiles via WMMA + streaming top-20.
// 2 waves / block: wave 0 = WMMA producer (A frags stay resident in VGPRs),
// wave 1 = top-k consumer with the per-lane sorted top-20 lists kept in LDS
// (only a scalar threshold lives in a register), 16 rows x even/odd column
// sub-streams, merged at the end. B staging overlaps the top-k scan.
// ---------------------------------------------------------------------------
__global__ void dist_topk_kernel(const float* __restrict__ x,
                                 const float* __restrict__ sq,
                                 int* __restrict__ idx) {
  extern __shared__ char smem[];
  _Float16* Asm = (_Float16*)smem;          // 16 x 64
  _Float16* Bsm = Asm + 1024;               // packed fragment layout, 1024 halves
  float*    Ssm = (float*)(Bsm + 1024);     // 16 x 17 score tile
  float*    Mv  = Ssm + 16 * 17;            // 32 x 20 sorted values (desc)
  int*      Mi  = (int*)(Mv + 32 * KK);     // 32 x 20 sorted indices
  const int b = blockIdx.y, t = blockIdx.x;
  const int tid = threadIdx.x, wave = tid >> 5, lane = tid & 31;
  const float* xb = x + (size_t)b * CC * NN;

  if (wave == 0) {
    for (int i = lane; i < 16 * 64; i += 32) {
      int m = i >> 6, c = i & 63;
      Asm[i] = (_Float16)xb[c * NN + t * 16 + m];
    }
  } else {
    for (int j = lane; j < 32 * KK; j += 32) { Mv[j] = -3.0e38f; Mi[j] = 0; }
  }
  __syncthreads();
  v16h a0 = {}, a1 = {};
  if (wave == 0) {
    a0 = frag_a2(Asm, 64, 1);        // K = 0..31  (loop invariant)
    a1 = frag_a2(Asm + 32, 64, 1);   // K = 32..63
  }

  const int row = lane & 15, half = lane >> 4;
  const int base = lane * KK;
  float sqrow = (wave == 1) ? sq[b * NN + t * 16 + row] : 0.f;
  float thr = -3.0e38f;                      // current 20th-best (quick reject)

  for (int jt = 0; jt < NN / 16; ++jt) {
    if (wave == 0) {
      // stage B (64K x 16N) in fragment-packed order
      for (int i = lane; i < 64 * 16; i += 32) {
        int c = i >> 4, n = i & 15;
        Bsm[(((c >> 4) * 16 + n) << 4) + (c & 15)] =
            (_Float16)xb[c * NN + jt * 16 + n];
      }
    }
    __syncthreads();                       // B ready; previous S consumed
    if (wave == 0) {
      v8f acc = {};
      acc = wmma32(a0, frag_b_packed(Bsm), acc);
      acc = wmma32(a1, frag_b_packed(Bsm + 512), acc);
      int n = lane & 15, grp = lane >> 4;
      float sqn = sq[b * NN + jt * 16 + n];
#pragma unroll
      for (int v = 0; v < 8; ++v)
        Ssm[(v + 8 * grp) * 17 + n] = 2.f * acc[v] - sqn;
    }
    __syncthreads();                       // S ready
    if (wave == 1) {
#pragma unroll
      for (int u = 0; u < 8; ++u) {        // even/odd sub-stream per lane
        int n = 2 * u + half;
        float v = Ssm[row * 17 + n] - sqrow;
        if (v > thr) {                     // rare path: LDS insertion sort
          int ci = jt * 16 + n;
          int j = KK - 1;
          while (j > 0) {
            float pv = Mv[base + j - 1];
            if (v > pv) {
              Mv[base + j] = pv; Mi[base + j] = Mi[base + j - 1]; --j;
            } else break;
          }
          Mv[base + j] = v; Mi[base + j] = ci;
          thr = Mv[base + KK - 1];
        }
      }
    }
  }
  __syncthreads();
  if (wave == 1 && lane < 16) {            // merge even/odd sorted lists
    const float* v0 = Mv + lane * KK;
    const float* v1 = Mv + (lane + 16) * KK;
    const int*   i0 = Mi + lane * KK;
    const int*   i1 = Mi + (lane + 16) * KK;
    int p0 = 0, p1 = 0;
    int* op = idx + ((size_t)b * NN + t * 16 + lane) * KK;
    for (int j = 0; j < KK; ++j) {
      if (v0[p0] >= v1[p1]) op[j] = i0[p0++];
      else                  op[j] = i1[p1++];
    }
  }
}

// ---------------------------------------------------------------------------
// Kernel 3: edge features -> e1 (W1) -> e2/delta (W2) and lf_pre (W4, f16 out)
// Block = 16 points, 128 threads (4 waves). ef/e1h in fragment-packed LDS.
// ---------------------------------------------------------------------------
__global__ void edge_conv_kernel(const float* __restrict__ x, const int* __restrict__ idx,
                                 const float* __restrict__ W1, const float* __restrict__ bn1,
                                 const float* __restrict__ W2, const float* __restrict__ bn2,
                                 const float* __restrict__ W4, const float* __restrict__ bn4,
                                 float* __restrict__ delta, _Float16* __restrict__ lf_pre) {
  extern __shared__ char smem[];
  _Float16* ef   = (_Float16*)smem;          // 128 x 320 (packed)
  _Float16* e1h  = ef + 128 * 320;           // 64 x 320 (packed)
  _Float16* wst  = e1h + 64 * 320;           // <= 64 x 128 weight stage
  int*      idxs = (int*)(wst + 64 * 128);   // 16*20
  const int tid = threadIdx.x, wave = tid >> 5, lane = tid & 31;
  const int b = blockIdx.y, t = blockIdx.x;
  const float* xb = x + (size_t)b * CC * NN;

  __builtin_prefetch(W4 + (size_t)tid * 1280, 0, 0);
  __builtin_prefetch(W2 + (size_t)tid * 640, 0, 0);

  for (int i = tid; i < 16 * KK; i += 128)
    idxs[i] = idx[((size_t)b * NN + t * 16) * KK + i];
  for (int i = tid; i < 64 * 128; i += 128) wst[i] = (_Float16)W1[i];
  __syncthreads();

  // build ef[c][col] packed, col = p*20+k : [nbr-center ; center]
  for (int i = tid; i < 128 * 320; i += 128) {
    int c = i / 320, col = i - c * 320;
    int p = col / KK, k = col - p * KK;
    int np = t * 16 + p, j = idxs[p * KK + k];
    float v = (c < 64) ? (xb[c * NN + j] - xb[c * NN + np]) : xb[(c - 64) * NN + np];
    ef[pkidx(c, col)] = (_Float16)v;
  }
  __syncthreads();

  // e1 = lrelu(bn1(W1 @ ef)) : nt-outer (B frags hoisted), mt-inner
  for (int nt = wave; nt < 20; nt += 4) {
    v16h bf[4];
#pragma unroll
    for (int ks = 0; ks < 4; ++ks) bf[ks] = frag_b_packed(ef + (ks * 20 + nt) * 512);
#pragma unroll
    for (int mt = 0; mt < 4; ++mt) {
      v8f acc = {};
#pragma unroll
      for (int ks = 0; ks < 4; ++ks)
        acc = wmma32(frag_a2(wst + mt * 16 * 128 + ks * 32, 128, 1), bf[ks], acc);
      int nloc = lane & 15, grp = lane >> 4;
#pragma unroll
      for (int v = 0; v < 8; ++v) {
        int o = mt * 16 + v + 8 * grp;
        e1h[pkidx(o, nt * 16 + nloc)] =
            (_Float16)d_lrelu(bn_apply(bn1, 64, o, acc[v]));
      }
    }
  }

  // lf_pre = lrelu(bn4(sum_k W4[:,:,k] @ ef_k)) : per-k 64x128 @ 128x16
  v8f accL = {};
  for (int k = 0; k < KK; ++k) {
    __syncthreads();
    for (int i = tid; i < 64 * 128; i += 128) wst[i] = (_Float16)W4[i * KK + k];
    __syncthreads();
#pragma unroll
    for (int ks = 0; ks < 4; ++ks)
      accL = wmma32(frag_a2(wst + wave * 16 * 128 + ks * 32, 128, 1),
                    frag_b_pkcol(ef, ks, k), accL);
  }
  {
    int nloc = lane & 15, grp = lane >> 4;
#pragma unroll
    for (int v = 0; v < 8; ++v) {
      int o = wave * 16 + v + 8 * grp;
      lf_pre[((size_t)b * 64 + o) * NN + t * 16 + nloc] =
          (_Float16)d_lrelu(bn_apply(bn4, 64, o, accL[v]));
    }
  }

  // e2 = lrelu(bn2(sum_k W2[:,:,k] @ e1_k)); delta = e2 - x
  v8f accE = {};
  for (int k = 0; k < KK; ++k) {
    __syncthreads();
    for (int i = tid; i < 64 * 64; i += 128) wst[i] = (_Float16)W2[i * KK + k];
    __syncthreads();
#pragma unroll
    for (int ks = 0; ks < 2; ++ks)
      accE = wmma32(frag_a2(wst + wave * 16 * 64 + ks * 32, 64, 1),
                    frag_b_pkcol(e1h, ks, k), accE);
  }
  {
    int nloc = lane & 15, grp = lane >> 4;
#pragma unroll
    for (int v = 0; v < 8; ++v) {
      int c = wave * 16 + v + 8 * grp;
      int np = t * 16 + nloc;
      float e2v = d_lrelu(bn_apply(bn2, 64, c, accE[v]));
      delta[((size_t)b * 64 + c) * NN + np] = e2v - xb[c * NN + np];
    }
  }
}

// ---------------------------------------------------------------------------
// Kernel 4: gf_pre = max_k( e1 + lrelu(bn3(W3 @ df)) )   (f16 out)
// Recomputes e1 from (x, idx1); df from (delta, idx2). ef packed.
// ---------------------------------------------------------------------------
__global__ void gf_max_kernel(const float* __restrict__ x, const float* __restrict__ delta,
                              const int* __restrict__ idx1, const int* __restrict__ idx2,
                              const float* __restrict__ W1, const float* __restrict__ bn1,
                              const float* __restrict__ W3, const float* __restrict__ bn3,
                              _Float16* __restrict__ gf_pre) {
  extern __shared__ char smem[];
  _Float16* ef   = (_Float16*)smem;           // 128 x 320 packed (ef then df)
  float*    sbuf = (float*)(ef + 128 * 320);  // 64 x 320 linear (e1 then e1+r)
  _Float16* wst  = (_Float16*)(sbuf + 64 * 320);
  int*      idxs = (int*)(wst + 64 * 128);    // 2 x 320
  const int tid = threadIdx.x, wave = tid >> 5, lane = tid & 31;
  const int b = blockIdx.y, t = blockIdx.x;
  const float* xb = x + (size_t)b * CC * NN;
  const float* db = delta + (size_t)b * CC * NN;

  for (int i = tid; i < 16 * KK; i += 128) {
    idxs[i]       = idx1[((size_t)b * NN + t * 16) * KK + i];
    idxs[320 + i] = idx2[((size_t)b * NN + t * 16) * KK + i];
  }
  for (int i = tid; i < 64 * 128; i += 128) wst[i] = (_Float16)W1[i];
  __syncthreads();

  // phase 1: ef from x/idx1, sbuf = lrelu(bn1(W1 @ ef))
  for (int i = tid; i < 128 * 320; i += 128) {
    int c = i / 320, col = i - c * 320;
    int p = col / KK, k = col - p * KK;
    int np = t * 16 + p, j = idxs[p * KK + k];
    float v = (c < 64) ? (xb[c * NN + j] - xb[c * NN + np]) : xb[(c - 64) * NN + np];
    ef[pkidx(c, col)] = (_Float16)v;
  }
  __syncthreads();
  for (int nt = wave; nt < 20; nt += 4) {
    v16h bf[4];
#pragma unroll
    for (int ks = 0; ks < 4; ++ks) bf[ks] = frag_b_packed(ef + (ks * 20 + nt) * 512);
#pragma unroll
    for (int mt = 0; mt < 4; ++mt) {
      v8f acc = {};
#pragma unroll
      for (int ks = 0; ks < 4; ++ks)
        acc = wmma32(frag_a2(wst + mt * 16 * 128 + ks * 32, 128, 1), bf[ks], acc);
      int nloc = lane & 15, grp = lane >> 4;
#pragma unroll
      for (int v = 0; v < 8; ++v) {
        int o = mt * 16 + v + 8 * grp;
        sbuf[o * 320 + nt * 16 + nloc] = d_lrelu(bn_apply(bn1, 64, o, acc[v]));
      }
    }
  }
  __syncthreads();

  // phase 2: df from delta/idx2, sbuf += lrelu(bn3(W3 @ df))
  for (int i = tid; i < 128 * 320; i += 128) {
    int c = i / 320, col = i - c * 320;
    int p = col / KK, k = col - p * KK;
    int np = t * 16 + p, j = idxs[320 + p * KK + k];
    float v = (c < 64) ? (db[c * NN + j] - db[c * NN + np]) : db[(c - 64) * NN + np];
    ef[pkidx(c, col)] = (_Float16)v;
  }
  for (int i = tid; i < 64 * 128; i += 128) wst[i] = (_Float16)W3[i];
  __syncthreads();
  for (int nt = wave; nt < 20; nt += 4) {
    v16h bf[4];
#pragma unroll
    for (int ks = 0; ks < 4; ++ks) bf[ks] = frag_b_packed(ef + (ks * 20 + nt) * 512);
#pragma unroll
    for (int mt = 0; mt < 4; ++mt) {
      v8f acc = {};
#pragma unroll
      for (int ks = 0; ks < 4; ++ks)
        acc = wmma32(frag_a2(wst + mt * 16 * 128 + ks * 32, 128, 1), bf[ks], acc);
      int nloc = lane & 15, grp = lane >> 4;
#pragma unroll
      for (int v = 0; v < 8; ++v) {
        int o = mt * 16 + v + 8 * grp;
        int s = o * 320 + nt * 16 + nloc;
        sbuf[s] = sbuf[s] + d_lrelu(bn_apply(bn3, 64, o, acc[v]));
      }
    }
  }
  __syncthreads();

  // phase 3: max over k
  for (int i = tid; i < 64 * 16; i += 128) {
    int c = i >> 4, p = i & 15;
    float m = -3.0e38f;
#pragma unroll
    for (int k = 0; k < KK; ++k) m = fmaxf(m, sbuf[c * 320 + p * KK + k]);
    gf_pre[((size_t)b * 64 + c) * NN + t * 16 + p] = (_Float16)m;
  }
}

// ---------------------------------------------------------------------------
// Kernel 5: channel attention part A: q/k GEMMs (K=1024), scores, softmax -> w
// One block per batch, 256 threads (8 waves). X staged via TDM (f16 raw copy).
// q/k results kept in fragment-packed LDS for the scores GEMM.
// ---------------------------------------------------------------------------
__global__ void attn_qk_kernel(const _Float16* __restrict__ X,
                               const float* __restrict__ Wq, const float* __restrict__ bnq,
                               const float* __restrict__ Wk, const float* __restrict__ bnk,
                               float* __restrict__ wout) {
  extern __shared__ char smem[];
  _Float16* Xs  = (_Float16*)smem;       // 64 x 1024
  _Float16* qs  = Xs + 64 * 1024;        // 128 x 64 (packed)
  _Float16* ksm = qs + 128 * 64;         // 128 x 64 (packed)
  float*    ssm = (float*)(ksm + 128 * 64);  // 64 x 68
  const int tid = threadIdx.x, wave = tid >> 5, lane = tid & 31;
  const int b = blockIdx.x;

  __builtin_prefetch(Wq + (size_t)tid * 512, 0, 0);
  __builtin_prefetch(Wk + (size_t)tid * 512, 0, 0);

  stage_lds_copy(smem, Xs, X + (size_t)b * 64 * NN, 64 * NN * 2);
  __syncthreads();

  // q[h,c] = relu(bn(sum_n W[h,n] X[c,n])) ; same for k
  for (int job = wave; job < 64; job += 8) {
    int use_k = job >> 5;
    int jj = job & 31;
    int mt = jj & 7, nt = jj >> 3;
    const float* Wg  = use_k ? Wk  : Wq;
    const float* bng = use_k ? bnk : bnq;
    _Float16*    dst = use_k ? ksm : qs;
    v8f acc = {};
    for (int kc = 0; kc < 32; ++kc)
      acc = wmma32(frag_a_f32(Wg + mt * 16 * NN + kc * 32, NN),
                   frag_b2(Xs + kc * 32, 1, NN), acc);
    int nloc = lane & 15, grp = lane >> 4;
#pragma unroll
    for (int v = 0; v < 8; ++v) {
      int h = mt * 16 + v + 8 * grp;
      dst[qpk(h, nt * 16 + nloc)] = (_Float16)d_relu(bn_apply(bng, HH, h, acc[v]));
    }
  }
  __syncthreads();

  // scores[c,d] = sum_h k[h,c] q[h,d]
  for (int job = wave; job < 16; job += 8) {
    int mt = job & 3, nt = job >> 2;
    v8f acc = {};
#pragma unroll
    for (int ks = 0; ks < 4; ++ks)
      acc = wmma32(frag_a_qpk(ksm, ks * 32, mt), frag_b_qpk(qs, ks * 32, nt), acc);
    int nloc = lane & 15, grp = lane >> 4;
#pragma unroll
    for (int v = 0; v < 8; ++v)
      ssm[(mt * 16 + v + 8 * grp) * 68 + nt * 16 + nloc] = acc[v];
  }
  __syncthreads();

  // w = softmax(mx - scores) along d
  if (tid < 64) {
    float mx = -3.0e38f;
    for (int d = 0; d < 64; ++d) mx = fmaxf(mx, ssm[tid * 68 + d]);
    float m2 = -3.0e38f;
    for (int d = 0; d < 64; ++d) m2 = fmaxf(m2, mx - ssm[tid * 68 + d]);
    float sum = 0.f;
    float* wrow = wout + ((size_t)b * 64 + tid) * 64;
    for (int d = 0; d < 64; ++d) {
      float e = expf(mx - ssm[tid * 68 + d] - m2);
      wrow[d] = e; sum += e;
    }
    float inv = 1.f / sum;
    for (int d = 0; d < 64; ++d) wrow[d] *= inv;
  }
}

// ---------------------------------------------------------------------------
// Kernel 6: channel attention part B: v = relu(bnv(Wv @ X)); out = a*w@v + X
// Fused per 16-column tile; per-wave packed v tile in LDS. X staged via TDM.
// ---------------------------------------------------------------------------
__global__ void attn_out_kernel(const _Float16* __restrict__ X,
                                const float* __restrict__ Wv, const float* __restrict__ bnv,
                                const float* __restrict__ alpha_p,
                                const float* __restrict__ wbuf,
                                float* __restrict__ out) {
  extern __shared__ char smem[];
  _Float16* Xs  = (_Float16*)smem;     // 64 x 1024
  _Float16* wvs = Xs + 64 * 1024;      // 64 x 64
  _Float16* wsm = wvs + 64 * 64;       // 64 x 64
  _Float16* vts = wsm + 64 * 64;       // 8 waves x (64 x 16, packed)
  const int tid = threadIdx.x, wave = tid >> 5, lane = tid & 31;
  const int b = blockIdx.x;
  const float alpha = alpha_p[0];

  stage_lds_copy(smem, Xs, X + (size_t)b * 64 * NN, 64 * NN * 2);
  for (int i = tid; i < 64 * 64; i += 256) {
    wvs[i] = (_Float16)Wv[i];
    wsm[i] = (_Float16)wbuf[(size_t)b * 4096 + i];
  }
  __syncthreads();

  _Float16* vt = vts + wave * 64 * 16;
  for (int nt = wave; nt < 64; nt += 8) {   // uniform: 8 iterations per wave
    v16h xb0 = frag_b2(Xs + nt * 16, NN, 1);            // K rows 0..31
    v16h xb1 = frag_b2(Xs + 32 * NN + nt * 16, NN, 1);  // K rows 32..63
#pragma unroll
    for (int mt = 0; mt < 4; ++mt) {
      v8f acc = {};
      acc = wmma32(frag_a2(wvs + mt * 16 * 64, 64, 1), xb0, acc);
      acc = wmma32(frag_a2(wvs + mt * 16 * 64 + 32, 64, 1), xb1, acc);
      int nloc = lane & 15, grp = lane >> 4;
#pragma unroll
      for (int v = 0; v < 8; ++v) {
        int d = mt * 16 + v + 8 * grp;
        // packed: (d>>4)*256 + n*16 + (d&15)
        vt[(d >> 4) * 256 + nloc * 16 + (d & 15)] =
            (_Float16)d_relu(bn_apply(bnv, 64, d, acc[v]));
      }
    }
    __syncthreads();
    v16h vb0 = frag_b_packed(vt);
    v16h vb1 = frag_b_packed(vt + 512);
#pragma unroll
    for (int mt = 0; mt < 4; ++mt) {
      v8f acc = {};
      acc = wmma32(frag_a2(wsm + mt * 16 * 64, 64, 1), vb0, acc);
      acc = wmma32(frag_a2(wsm + mt * 16 * 64 + 32, 64, 1), vb1, acc);
      int nloc = lane & 15, grp = lane >> 4;
#pragma unroll
      for (int v = 0; v < 8; ++v) {
        int c = mt * 16 + v + 8 * grp;
        float xv = (float)Xs[c * NN + nt * 16 + nloc];
        out[((size_t)b * 64 + c) * NN + nt * 16 + nloc] = alpha * acc[v] + xv;
      }
    }
    __syncthreads();
  }
}

// ---------------------------------------------------------------------------
// Host launcher
// ---------------------------------------------------------------------------
extern "C" void kernel_launch(void* const* d_in, const int* in_sizes, int n_in,
                              void* d_out, int out_size, void* d_ws, size_t ws_size,
                              hipStream_t stream) {
  (void)in_sizes; (void)n_in; (void)out_size; (void)ws_size;
  const float* x      = (const float*)d_in[0];
  const float* W1     = (const float*)d_in[1];
  const float* bn1    = (const float*)d_in[2];
  const float* W2     = (const float*)d_in[3];
  const float* bn2    = (const float*)d_in[4];
  const float* W3     = (const float*)d_in[5];
  const float* bn3    = (const float*)d_in[6];
  const float* W4     = (const float*)d_in[7];
  const float* bn4    = (const float*)d_in[8];
  const float* a1_Wq  = (const float*)d_in[9];
  const float* a1_bnq = (const float*)d_in[10];
  const float* a1_Wk  = (const float*)d_in[11];
  const float* a1_bnk = (const float*)d_in[12];
  const float* a1_Wv  = (const float*)d_in[13];
  const float* a1_bnv = (const float*)d_in[14];
  const float* a1_al  = (const float*)d_in[15];
  const float* a2_Wq  = (const float*)d_in[16];
  const float* a2_bnq = (const float*)d_in[17];
  const float* a2_Wk  = (const float*)d_in[18];
  const float* a2_bnk = (const float*)d_in[19];
  const float* a2_Wv  = (const float*)d_in[20];
  const float* a2_bnv = (const float*)d_in[21];
  const float* a2_al  = (const float*)d_in[22];

  char* ws = (char*)d_ws;
  float*     sq     = (float*)(ws + 0);             //  64 KB
  int*       idx1   = (int*)(ws + 65536);           // 1.25 MB
  int*       idx2   = (int*)(ws + 1376256);         // 1.25 MB
  float*     delta  = (float*)(ws + 2686976);       // 4 MB
  _Float16*  lf_pre = (_Float16*)(ws + 6881280);    // 2 MB (f16)
  _Float16*  gf_pre = (_Float16*)(ws + 8978432);    // 2 MB (f16)
  float*     wbuf1  = (float*)(ws + 11075584);      // 256 KB
  float*     wbuf2  = (float*)(ws + 11337728);      // 256 KB
  float* out_gf = (float*)d_out;
  float* out_lf = out_gf + (size_t)BB * 64 * NN;

  constexpr unsigned DIST_SHM = 1024 * 2 + 1024 * 2 + 16 * 17 * 4 +
                                32 * KK * 4 + 32 * KK * 4;                         //  10304
  constexpr unsigned EC_SHM   = 128 * 320 * 2 + 64 * 320 * 2 + 64 * 128 * 2 + 1280; // 140544
  constexpr unsigned GF_SHM   = 128 * 320 * 2 + 64 * 320 * 4 + 64 * 128 * 2 + 2560; // 182784
  constexpr unsigned QK_SHM   = 64 * 1024 * 2 + 2 * 128 * 64 * 2 + 64 * 68 * 4;     // 181248
  constexpr unsigned OUT_SHM  = 64 * 1024 * 2 + 2 * 64 * 64 * 2 + 8 * 64 * 16 * 2;  // 163840

  // CDNA5 allows up to 320 KB LDS per workgroup; raise the dynamic-LDS caps.
  (void)hipFuncSetAttribute((const void*)edge_conv_kernel,
                            hipFuncAttributeMaxDynamicSharedMemorySize, EC_SHM);
  (void)hipFuncSetAttribute((const void*)gf_max_kernel,
                            hipFuncAttributeMaxDynamicSharedMemorySize, GF_SHM);
  (void)hipFuncSetAttribute((const void*)attn_qk_kernel,
                            hipFuncAttributeMaxDynamicSharedMemorySize, QK_SHM);
  (void)hipFuncSetAttribute((const void*)attn_out_kernel,
                            hipFuncAttributeMaxDynamicSharedMemorySize, OUT_SHM);

  sqnorm_kernel<<<BB * NN / 256, 256, 0, stream>>>(x, sq);
  dist_topk_kernel<<<dim3(NN / 16, BB), 64, DIST_SHM, stream>>>(x, sq, idx1);
  edge_conv_kernel<<<dim3(NN / 16, BB), 128, EC_SHM, stream>>>(
      x, idx1, W1, bn1, W2, bn2, W4, bn4, delta, lf_pre);
  sqnorm_kernel<<<BB * NN / 256, 256, 0, stream>>>(delta, sq);
  dist_topk_kernel<<<dim3(NN / 16, BB), 64, DIST_SHM, stream>>>(delta, sq, idx2);
  gf_max_kernel<<<dim3(NN / 16, BB), 128, GF_SHM, stream>>>(
      x, delta, idx1, idx2, W1, bn1, W3, bn3, gf_pre);
  attn_qk_kernel<<<BB, 256, QK_SHM, stream>>>(gf_pre, a1_Wq, a1_bnq, a1_Wk, a1_bnk, wbuf1);
  attn_out_kernel<<<BB, 256, OUT_SHM, stream>>>(gf_pre, a1_Wv, a1_bnv, a1_al, wbuf1, out_gf);
  attn_qk_kernel<<<BB, 256, QK_SHM, stream>>>(lf_pre, a2_Wq, a2_bnq, a2_Wk, a2_bnk, wbuf2);
  attn_out_kernel<<<BB, 256, OUT_SHM, stream>>>(lf_pre, a2_Wv, a2_bnv, a2_al, wbuf2, out_lf);
}
